// RVQQuantizer_50070728737555
// MI455X (gfx1250) — compile-verified, matrix operand order checked by reading, not run on hardware
//
#include <hip/hip_runtime.h>
#include <hip/hip_bf16.h>
#include <cstdint>

// ---- problem constants (match reference setup_inputs) ----
#define BT   8192        // B*T = 8*1024
#define DDIM 256         // D
#define KC   4096        // codebook size K
#define EPSV 1e-10f

typedef __attribute__((ext_vector_type(16))) __bf16 v16bf;
typedef __attribute__((ext_vector_type(4)))  __bf16 v4bf;
typedef __attribute__((ext_vector_type(8)))  float  v8f;

// ---------------- prep: f32 -> bf16 (z) ----------------
__global__ __launch_bounds__(256) void conv_z_kernel(const float* __restrict__ in,
                                                     __bf16* __restrict__ out, int n4) {
  int i = blockIdx.x * 256 + threadIdx.x;
  if (i < n4) {
    float4 f = ((const float4*)in)[i];
    v4bf o;
    o[0] = (__bf16)f.x; o[1] = (__bf16)f.y; o[2] = (__bf16)f.z; o[3] = (__bf16)f.w;
    ((v4bf*)out)[i] = o;
  }
}

// ------- prep: codebook f32 -> bf16 + e_sq (one wave per row) -------
__global__ __launch_bounds__(256) void cb_prep_kernel(const float* __restrict__ cb,
                                                      __bf16* __restrict__ cbb,
                                                      float* __restrict__ esq) {
  int lane = threadIdx.x & 31;
  int row  = blockIdx.x * 8 + (threadIdx.x >> 5);
  const float* r = cb + (size_t)row * DDIM;
  float s = 0.f;
  #pragma unroll
  for (int j = lane; j < DDIM; j += 32) {
    float v = r[j];
    s += v * v;
    cbb[(size_t)row * DDIM + j] = (__bf16)v;
  }
  #pragma unroll
  for (int m = 16; m >= 1; m >>= 1) s += __shfl_xor(s, m, 32);
  if (lane == 0) esq[row] = s;
}

// ---------------- prep: zero histogram + commitment slot ----------------
__global__ __launch_bounds__(256) void init_kernel(float* __restrict__ n_i,
                                                   float* __restrict__ commit_slot) {
  int i = blockIdx.x * 256 + threadIdx.x;
  if (i < KC) n_i[i] = 0.f;
  if (i == 0) *commit_slot = 0.f;
}

// ---------------- main fused WMMA scoring + gumbel argmax ----------------
// block = 128 threads = 4 waves; each wave owns 16 token rows, scans all K codes.
// Per iteration: 32 codes = 2 N-tiles -> 2 independent WMMA accumulator chains,
// A-fragments (held in VGPRs for the whole kernel) reused by both chains.
__global__ __launch_bounds__(128) void vq_score_argmax_kernel(
    const float*  __restrict__ u,       // [BT][KC] f32, streamed once (non-temporal)
    const __bf16* __restrict__ zb,      // [BT][DDIM] bf16 row-major
    const __bf16* __restrict__ cbb,     // [KC][DDIM] bf16 row-major (L2-resident)
    const float*  __restrict__ esq,     // [KC]
    int*          __restrict__ idx_out, // [BT]
    float*        __restrict__ n_i)     // [KC] histogram (pre-zeroed)
{
  const int lane    = threadIdx.x & 31;
  const int wave    = threadIdx.x >> 5;
  const int rowBase = blockIdx.x * 64 + wave * 16;
  const int half    = lane >> 4;      // 0: K 0-15 / rows M 0-7 ; 1: K 16-31 / rows M 8-15
  const int ln      = lane & 15;

  // A-fragments: 16x32 bf16 per k-step; lane = row, 16 contiguous dims per lane (32B).
  v16bf afrag[8];
  const __bf16* zrow = zb + (size_t)(rowBase + ln) * DDIM + half * 16;
  #pragma unroll
  for (int ks = 0; ks < 8; ++ks)
    afrag[ks] = *(const v16bf*)(zrow + ks * 32);

  float bestv[8];
  int   besti[8];
  #pragma unroll
  for (int r = 0; r < 8; ++r) { bestv[r] = -3.4e38f; besti[r] = 0; }

  for (int nb = 0; nb < KC; nb += 32) {
    // B-fragments: 32x16 bf16; lane column = code, 16 contiguous dims per lane (32B).
    const __bf16* crow0 = cbb + (size_t)(nb + ln) * DDIM + half * 16;       // codes nb..nb+15
    const __bf16* crow1 = crow0 + (size_t)16 * DDIM;                         // codes nb+16..nb+31

    // prefetch next block's codebook rows for this lane (global_prefetch_b8)
    if (nb + 32 < KC)
      __builtin_prefetch(crow0 + (size_t)32 * DDIM, 0, 3);

    v8f acc0 = {0.f, 0.f, 0.f, 0.f, 0.f, 0.f, 0.f, 0.f};
    v8f acc1 = {0.f, 0.f, 0.f, 0.f, 0.f, 0.f, 0.f, 0.f};
    #pragma unroll
    for (int ks = 0; ks < 8; ++ks) {
      v16bf b0 = *(const v16bf*)(crow0 + ks * 32);
      v16bf b1 = *(const v16bf*)(crow1 + ks * 32);
      acc0 = __builtin_amdgcn_wmma_f32_16x16x32_bf16(
          false, afrag[ks], false, b0, (short)0, acc0, false, false);
      acc1 = __builtin_amdgcn_wmma_f32_16x16x32_bf16(
          false, afrag[ks], false, b1, (short)0, acc1, false, false);
    }

    const int   ng0 = nb + ln;        // global code index, tile 0 column
    const int   ng1 = ng0 + 16;       // tile 1 column
    const float es0 = esq[ng0];
    const float es1 = esq[ng1];
    #pragma unroll
    for (int r = 0; r < 8; ++r) {
      const int m = rowBase + r + 8 * half;                  // token row of acc[r]
      const float* urow = u + (size_t)m * KC;
      float u0 = __builtin_nontemporal_load(urow + ng0);
      float u1 = __builtin_nontemporal_load(urow + ng1);
      float g0 = -__logf(-__logf(u0 + EPSV) + EPSV);         // gumbel noise
      float g1 = -__logf(-__logf(u1 + EPSV) + EPSV);
      float v0 = 2.0f * acc0[r] - es0 + g0;                  // logits + gumbel (z_sq dropped)
      float v1 = 2.0f * acc1[r] - es1 + g1;
      if (v0 > bestv[r] || (v0 == bestv[r] && ng0 < besti[r])) { bestv[r] = v0; besti[r] = ng0; }
      if (v1 > bestv[r] || (v1 == bestv[r] && ng1 < besti[r])) { bestv[r] = v1; besti[r] = ng1; }
    }
  }

  // reduce (val,idx) across the 16 lanes holding the same row (xor masks stay in half)
  #pragma unroll
  for (int r = 0; r < 8; ++r) {
    float v = bestv[r];
    int   i = besti[r];
    #pragma unroll
    for (int m = 8; m >= 1; m >>= 1) {
      float ov = __shfl_xor(v, m, 32);
      int   oi = __shfl_xor(i, m, 32);
      if (ov > v || (ov == v && oi < i)) { v = ov; i = oi; }
    }
    if (ln == 0) {
      int m = rowBase + r + 8 * half;
      idx_out[m] = i;
      atomicAdd(&n_i[i], 1.0f);
    }
  }
}

// ---------------- gather embedding / z_q + commitment ----------------
__global__ __launch_bounds__(256) void gather_kernel(
    const float* __restrict__ z, const float* __restrict__ cb,
    const int*  __restrict__ idx,
    float* __restrict__ zq, float* __restrict__ emb,
    float* __restrict__ idxf, float* __restrict__ commit)
{
  __shared__ float red[256];
  const int m = blockIdx.x;
  const int d = threadIdx.x;
  const int k = idx[m];
  float e  = cb[(size_t)k * DDIM + d];
  float zv = z [(size_t)m * DDIM + d];
  zq [(size_t)m * DDIM + d] = e;   // z + sg(quantized - z) == quantized (one-hot codes)
  emb[(size_t)m * DDIM + d] = e;
  float diff = e - zv;
  red[d] = diff * diff;
  __syncthreads();
  #pragma unroll
  for (int s = 128; s > 0; s >>= 1) {
    if (d < s) red[d] += red[d + s];
    __syncthreads();
  }
  if (d == 0) {
    atomicAdd(commit, red[0] * (1.0f / ((float)BT * (float)DDIM)));
    idxf[m] = (float)k;            // indices emitted as float in d_out
  }
}

// ---------------- usage EMA -> entropy / perplexity (1 workgroup) ----------------
__global__ __launch_bounds__(1024) void usage_kernel(
    const float* __restrict__ usage_in, const float* __restrict__ n_i,
    float* __restrict__ out_perp, float* __restrict__ out_ent)
{
  __shared__ float us[KC];
  __shared__ float red[1024];
  const int t = threadIdx.x;
  float s = 0.f;
  for (int i = t; i < KC; i += 1024) {
    float v = usage_in[i] * 0.99f + n_i[i];
    us[i] = v;
    s += v;
  }
  red[t] = s;
  __syncthreads();
  #pragma unroll
  for (int st = 512; st > 0; st >>= 1) {
    if (t < st) red[t] += red[t + st];
    __syncthreads();
  }
  const float total = red[0];
  __syncthreads();
  float ent = 0.f;
  for (int i = t; i < KC; i += 1024) {
    float p = (total > 0.f) ? (us[i] / (total + EPSV)) : (1.0f / (float)KC);
    ent += -p * __logf(p + EPSV);
  }
  red[t] = ent;
  __syncthreads();
  #pragma unroll
  for (int st = 512; st > 0; st >>= 1) {
    if (t < st) red[t] += red[t + st];
    __syncthreads();
  }
  if (t == 0) {
    *out_ent  = red[0];
    *out_perp = __expf(red[0]);
  }
}

extern "C" void kernel_launch(void* const* d_in, const int* in_sizes, int n_in,
                              void* d_out, int out_size, void* d_ws, size_t ws_size,
                              hipStream_t stream) {
  const float* z     = (const float*)d_in[0];   // [8,1024,256]
  const float* u     = (const float*)d_in[1];   // [8,1024,4096]
  const float* cb    = (const float*)d_in[2];   // [4096,256]
  const float* usage = (const float*)d_in[3];   // [4096]

  float* out = (float*)d_out;
  float* zq     = out;                               // 2,097,152
  float* emb    = out + (size_t)BT * DDIM;           // 2,097,152
  float* idxf   = out + (size_t)2 * BT * DDIM;       // 8,192
  float* commit = idxf + BT;                         // scalar
  float* perp   = commit + 1;                        // scalar
  float* ent    = perp + 1;                          // scalar

  // workspace layout
  uint8_t* ws = (uint8_t*)d_ws;
  __bf16* z_bf  = (__bf16*)(ws);                                  // 4 MB
  __bf16* cb_bf = (__bf16*)(ws + (size_t)BT * DDIM * 2);          // 2 MB
  float*  esq   = (float*) (ws + (size_t)BT * DDIM * 2 + (size_t)KC * DDIM * 2);
  float*  n_i   = esq + KC;
  int*    idx_i = (int*)(n_i + KC);

  // 1) prep
  conv_z_kernel<<<(BT * DDIM / 4 + 255) / 256, 256, 0, stream>>>(z, z_bf, BT * DDIM / 4);
  cb_prep_kernel<<<KC / 8, 256, 0, stream>>>(cb, cb_bf, esq);
  init_kernel<<<KC / 256, 256, 0, stream>>>(n_i, commit);

  // 2) fused WMMA scoring + gumbel argmax (64 rows/block, 32 codes/iter)
  vq_score_argmax_kernel<<<BT / 64, 128, 0, stream>>>(u, z_bf, cb_bf, esq, idx_i, n_i);

  // 3) gather + commitment
  gather_kernel<<<BT, 256, 0, stream>>>(z, cb, idx_i, zq, emb, idxf, commit);

  // 4) usage entropy / perplexity
  usage_kernel<<<1, 1024, 0, stream>>>(usage, n_i, perp, ent);
}
